// StateExtract_42623255445876
// MI455X (gfx1250) — matrix operand order repaired and checked by reading, not compiled, not run on hardware
//
#include <hip/hip_runtime.h>
#include <math.h>

// ---------------- CDNA5 WMMA operand types (wave32) ----------------
typedef __attribute__((ext_vector_type(16))) _Float16 v16h;
typedef __attribute__((ext_vector_type(8)))  _Float16 v8h;
typedef __attribute__((ext_vector_type(8)))  float    v8f;

// ---------------- problem sizes (match reference) ----------------
#define N_OP 50000
#define N_M  10000
#define N_A  10000
#define BSZ  128
#define F_OPF 32
#define F_MF  16
#define F_AF  16
#define F_GF  16
#define EDIM  8
#define CH    256
#define NL    3
#define E_PRED 100000
#define E_SUCC 100000
#define E_OM   200000
#define E_MA   100000

// ---------------- input / param leaf indices ----------------
// Top-level: setup_inputs() insertion order. params (idx 14) flattened as a
// JAX pytree: dict keys sorted (ASCII: 'W*' < lowercase).
enum {
  IN_X_OP=0, IN_X_M, IN_X_A, IN_EI_PRED, IN_EI_SUCC, IN_EI_OM_SRC, IN_EI_OM_DST,
  IN_EA_OM, IN_EI_MA_SRC, IN_EI_MA_DST, IN_GLOBAL, IN_BATCH_OP, IN_BATCH_M, IN_BATCH_A,
  P_CONV_A  = 14,   // Wl,Wr,att,bias,bl,br
  P_CONV_M  = 20,
  P_CONV_OP = 26,
  P_LAYER0  = 32,   // 31 leaves per layer
  P_MIX0    = 125,  // W1,W2,Wp,b1,b2,bp
  P_MIX1    = 131,  // W1,W2,b1,b2
  P_MIX2    = 135,  // W1,W2,Wp,b1,b2,bp
  P_MIXBN0  = 141,  // b,g
  P_MIXBN1  = 143,  // b,g
  P_PROJ_A  = 145,  // W,b
  P_PROJ_M  = 147,
  P_PROJ_OP = 149,
  P_TOK_A   = 151, P_TOK_M = 152, P_TOK_OP = 153,
  N_LEAVES  = 154
};
// per-layer leaf offsets (keys sorted: bn_a, bn_m, bn_op, ma, om, pred, succ)
#define L_BN_A_B  0
#define L_BN_A_G  1
#define L_BN_M_B  2
#define L_BN_M_G  3
#define L_BN_OP_B 4
#define L_BN_OP_G 5
#define L_MA      6   // Wl,Wr,att,bias,bl,br
#define L_OM      12  // We,Wl,Wr,att,bias,bl,br
#define L_PRED    19  // Wl,Wr,att,bias,bl,br
#define L_SUCC    25  // Wl,Wr,att,bias,bl,br
#define L_STRIDE  31

// =================================================================
// Utility kernels
// =================================================================

__global__ void fill_kernel(float* __restrict__ p, float v, long long n) {
  long long i = (long long)blockIdx.x * blockDim.x + threadIdx.x;
  if (i < n) p[i] = v;
}

__global__ void copy_kernel(float* __restrict__ dst, const float* __restrict__ src, long long n) {
  long long i = (long long)blockIdx.x * blockDim.x + threadIdx.x;
  if (i < n) dst[i] = src[i];
}

__global__ void add_inplace_kernel(float* __restrict__ y, const float* __restrict__ x, long long n) {
  long long i = (long long)blockIdx.x * blockDim.x + threadIdx.x;
  if (i < n) y[i] += x[i];
}

// f[i] = f[i] + (h ? h[i] : 0) + b1[c] + (b2 ? b2[c] : 0)
__global__ void bias_res_kernel(float* __restrict__ f, const float* __restrict__ h,
                                const float* __restrict__ b1, const float* __restrict__ b2,
                                int N, int Cc) {
  long long i = (long long)blockIdx.x * blockDim.x + threadIdx.x;
  if (i >= (long long)N * Cc) return;
  int c = (int)(i % Cc);
  float v = f[i] + b1[c];
  if (h)  v += h[i];
  if (b2) v += b2[c];
  f[i] = v;
}

// =================================================================
// WMMA GEMM: Y[N,Cc] = X[N,K] @ W[K,Cc] + bias, optional tanh / accumulate.
// Block = 128 threads (4 waves), block tile 64x64, K-step 32.
// Branch-free staging: clamped float4 global loads (global_load_b128) +
// value masking via selects; f32->f16 once into LDS; A row-major (stride 40
// halves), B transposed [col][k] so each fragment is two ds_load_b128 in the
// exact wave32 WMMA operand layout. 4 v_wmma per K-step per wave.
// Requires: Cc % 64 == 0, K % 4 == 0, K >= 4.
// =================================================================
#define LDSTR 40  // f16 stride (80B rows -> 16B-aligned b128 fragment loads)

__global__ __launch_bounds__(128)
void gemm_bias_kernel(const float* __restrict__ X, const float* __restrict__ Wt,
                      const float* __restrict__ bias, float* __restrict__ Y,
                      int N, int K, int Cc, int act, int accum) {
  __shared__ _Float16 As[64 * LDSTR];  // [row][k]
  __shared__ _Float16 Bs[64 * LDSTR];  // [col][k] (transposed during staging)

  int tilesN = Cc >> 6;
  int bm0 = (blockIdx.x / tilesN) * 64;
  int bn0 = (blockIdx.x % tilesN) * 64;

  int tid  = threadIdx.x;
  int lane = tid & 31;
  int wv   = tid >> 5;          // wave 0..3 -> row strip
  int half = lane >> 4;
  int l15  = lane & 15;

  v8f acc[4];
#pragma unroll
  for (int n = 0; n < 4; ++n) acc[n] = (v8f){0.f,0.f,0.f,0.f,0.f,0.f,0.f,0.f};

  for (int k0 = 0; k0 < K; k0 += 32) {
    // ---- stage A: 64 rows x 32 k (512 float4 / 128 threads) ----
#pragma unroll
    for (int it = 0; it < 4; ++it) {
      int q   = tid + it * 128;
      int row = q >> 3;              // 0..63
      int c4  = (q & 7) << 2;        // 0,4,..,28
      int grow = bm0 + row;
      int gr   = grow < N ? grow : (N - 1);
      int ks   = k0 + c4;
      int kc   = ks <= (K - 4) ? ks : (K - 4);
      const float4 v = *(const float4*)(X + (long long)gr * K + kc);
      bool rok = grow < N;
      _Float16* dst = &As[row * LDSTR + c4];
      dst[0] = (rok && (ks + 0) < K) ? (_Float16)v.x : (_Float16)0.f;
      dst[1] = (rok && (ks + 1) < K) ? (_Float16)v.y : (_Float16)0.f;
      dst[2] = (rok && (ks + 2) < K) ? (_Float16)v.z : (_Float16)0.f;
      dst[3] = (rok && (ks + 3) < K) ? (_Float16)v.w : (_Float16)0.f;
    }
    // ---- stage B (transposed): 32 k x 64 cols ----
#pragma unroll
    for (int it = 0; it < 4; ++it) {
      int q    = tid + it * 128;
      int krow = q >> 4;             // 0..31
      int c4   = (q & 15) << 2;      // 0,4,..,60
      int gk   = k0 + krow;
      int gkc  = gk < K ? gk : (K - 1);
      const float4 v = *(const float4*)(Wt + (long long)gkc * Cc + bn0 + c4);
      bool kok = gk < K;
      Bs[(c4 + 0) * LDSTR + krow] = kok ? (_Float16)v.x : (_Float16)0.f;
      Bs[(c4 + 1) * LDSTR + krow] = kok ? (_Float16)v.y : (_Float16)0.f;
      Bs[(c4 + 2) * LDSTR + krow] = kok ? (_Float16)v.z : (_Float16)0.f;
      Bs[(c4 + 3) * LDSTR + krow] = kok ? (_Float16)v.w : (_Float16)0.f;
    }
    __syncthreads();

    // ---- A fragment (wave32 layout): elems 0..7 -> k=8*half+e,
    //      elems 8..15 -> k=16+8*half+e ----
    int arow = 16 * wv + l15;
    v8h alo = *(const v8h*)&As[arow * LDSTR + half * 8];
    v8h ahi = *(const v8h*)&As[arow * LDSTR + 16 + half * 8];
    v16h a = __builtin_shufflevector(alo, ahi, 0,1,2,3,4,5,6,7,8,9,10,11,12,13,14,15);

#pragma unroll
    for (int n = 0; n < 4; ++n) {
      // ---- B fragment: col = 16n + l15, elems 0..7 -> k = 8*half + e, 8..15 -> k = 16+8*half+e ----
      int bcol = 16 * n + l15;
      v8h blo = *(const v8h*)&Bs[bcol * LDSTR + half * 8];
      v8h bhi = *(const v8h*)&Bs[bcol * LDSTR + 16 + half * 8];
      v16h b = __builtin_shufflevector(blo, bhi, 0,1,2,3,4,5,6,7,8,9,10,11,12,13,14,15);
      acc[n] = __builtin_amdgcn_wmma_f32_16x16x32_f16(false, a, false, b,
                                                      (short)0, acc[n], false, false);
    }
    __syncthreads();
  }

  // ---- epilogue: D VGPR i -> row = i + 8*half, col = l15 ----
#pragma unroll
  for (int n = 0; n < 4; ++n) {
    int col = bn0 + 16 * n + l15;
    float bv = bias ? bias[col] : 0.f;
#pragma unroll
    for (int i = 0; i < 8; ++i) {
      int r = bm0 + 16 * wv + i + 8 * half;
      if (r < N) {
        long long idx = (long long)r * Cc + col;
        float v = acc[n][i] + bv;
        if (accum) v += Y[idx];
        if (act == 1) v = tanhf(v);
        Y[idx] = v;
      }
    }
  }
}

// =================================================================
// GATv2 edge pipeline (wave32: one wave per edge, 8 channels/lane)
// =================================================================
__device__ inline void atomicMaxFloat(float* addr, float val) {
  unsigned int* ua = (unsigned int*)addr;
  unsigned int old = *ua;
  while (true) {
    float f = __uint_as_float(old);
    if (f >= val) break;
    unsigned int assumed = old;
    old = atomicCAS(ua, assumed, __float_as_uint(val));
    if (old == assumed) break;
  }
}

__global__ void edge_logits_kernel(const float* __restrict__ l, const float* __restrict__ r,
                                   int rstride,
                                   const int* __restrict__ src, const int* __restrict__ dst,
                                   const float* __restrict__ att,
                                   const float* __restrict__ ea, const float* __restrict__ We,
                                   float* __restrict__ evals, float* __restrict__ segmax,
                                   int E, int Cc) {
  int e = blockIdx.x * 8 + (threadIdx.x >> 5);
  int lane = threadIdx.x & 31;
  if (e >= E) return;
  int s = src ? src[e] : e;
  int d = dst[e];
  float eav[EDIM];
  if (ea) {
#pragma unroll
    for (int j = 0; j < EDIM; ++j) eav[j] = ea[(long long)e * EDIM + j];
  }
  float part = 0.f;
  for (int c = lane; c < Cc; c += 32) {
    float m = l[(long long)s * Cc + c] + r[(long long)d * rstride + c];
    if (ea) {
#pragma unroll
      for (int j = 0; j < EDIM; ++j) m += eav[j] * We[j * Cc + c];
    }
    m = (m > 0.f) ? m : 0.2f * m;
    part += att[c] * m;
  }
#pragma unroll
  for (int off = 16; off > 0; off >>= 1)
    part += __shfl_xor(part, off, 32);
  if (lane == 0) {
    evals[e] = part;
    atomicMaxFloat(&segmax[d], part);
  }
}

__global__ void edge_exp_kernel(const float* __restrict__ evals, const int* __restrict__ dst,
                                const float* __restrict__ segmax, float* __restrict__ segsum,
                                float* __restrict__ avals, int E) {
  int e = blockIdx.x * blockDim.x + threadIdx.x;
  if (e >= E) return;
  int d = dst[e];
  float ex = expf(evals[e] - segmax[d]);
  avals[e] = ex;
  atomicAdd(&segsum[d], ex);
}

__global__ void edge_scatter_kernel(const float* __restrict__ l,
                                    const int* __restrict__ src, const int* __restrict__ dst,
                                    const float* __restrict__ avals, const float* __restrict__ segsum,
                                    float* __restrict__ out, int E, int Cc) {
  int e = blockIdx.x * 8 + (threadIdx.x >> 5);
  int lane = threadIdx.x & 31;
  if (e >= E) return;
  int s = src ? src[e] : e;
  int d = dst[e];
  float coef = avals[e] / segsum[d];
  for (int c = lane; c < Cc; c += 32)
    atomicAdd(&out[(long long)d * Cc + c], coef * l[(long long)s * Cc + c]);
}

// =================================================================
// BatchNorm (over rows)
// =================================================================
__global__ void bn_stats_kernel(const float* __restrict__ x, float* __restrict__ sum,
                                float* __restrict__ sumsq, int N, int Cc) {
  int r0 = blockIdx.x * 16;
  int rend = r0 + 16; if (rend > N) rend = N;
  float s[4] = {0.f,0.f,0.f,0.f}, q[4] = {0.f,0.f,0.f,0.f};
  for (int r = r0; r < rend; ++r) {
    int j = 0;
    for (int c = threadIdx.x; c < Cc; c += 256, ++j) {
      float v = x[(long long)r * Cc + c];
      s[j] += v; q[j] += v * v;
    }
  }
  int j = 0;
  for (int c = threadIdx.x; c < Cc; c += 256, ++j) {
    atomicAdd(&sum[c], s[j]);
    atomicAdd(&sumsq[c], q[j]);
  }
}

__global__ void bn_apply_tanh_kernel(const float* __restrict__ x, float* __restrict__ y,
                                     const float* __restrict__ sum, const float* __restrict__ sumsq,
                                     const float* __restrict__ g, const float* __restrict__ b,
                                     int N, int Cc) {
  long long i = (long long)blockIdx.x * blockDim.x + threadIdx.x;
  if (i >= (long long)N * Cc) return;
  int c = (int)(i % Cc);
  float invN = 1.f / (float)N;
  float mu  = sum[c] * invN;
  float var = sumsq[c] * invN - mu * mu;
  float v = (x[i] - mu) * rsqrtf(var + 1e-5f) * g[c] + b[c];
  y[i] = tanhf(v);
}

// z = concat([global_attr(B,FG), g_op(B,C), g_m(B,C), g_a(B,C)], axis=1)
__global__ void concat_z_kernel(const float* __restrict__ ga, const float* __restrict__ gop,
                                const float* __restrict__ gm, const float* __restrict__ gaa,
                                float* __restrict__ z, int Bn, int FG, int Cc) {
  int zin = FG + 3 * Cc;
  long long i = (long long)blockIdx.x * blockDim.x + threadIdx.x;
  if (i >= (long long)Bn * zin) return;
  int row = (int)(i / zin), col = (int)(i % zin);
  float v;
  if      (col < FG)            v = ga [row * FG + col];
  else if (col < FG + Cc)       v = gop[row * Cc + (col - FG)];
  else if (col < FG + 2 * Cc)   v = gm [row * Cc + (col - FG - Cc)];
  else                          v = gaa[row * Cc + (col - FG - 2 * Cc)];
  z[i] = v;
}

// =================================================================
// Host-side launch helpers
// =================================================================
static inline int cdiv(long long a, long long b) { return (int)((a + b - 1) / b); }

static void lfill(hipStream_t s, float* p, float v, long long n) {
  fill_kernel<<<cdiv(n, 256), 256, 0, s>>>(p, v, n);
}

static void lgemm(hipStream_t s, const float* X, const float* W, const float* bias,
                  float* Y, int N, int K, int Cc, int act, int accum) {
  int tiles = cdiv(N, 64) * (Cc / 64);
  gemm_bias_kernel<<<tiles, 128, 0, s>>>(X, W, bias, Y, N, K, Cc, act, accum);
}

static void lgat(hipStream_t s, const float* l, const float* r, int rstride,
                 const int* src, const int* dst, const float* att,
                 const float* ea, const float* We,
                 float* evals, float* avals, float* segmax, float* segsum,
                 float* out, int E, int ndst, int Cc) {
  lfill(s, segmax, -3.0e38f, ndst);
  lfill(s, segsum, 0.f, ndst);
  int eb = cdiv(E, 8);
  edge_logits_kernel <<<eb, 256, 0, s>>>(l, r, rstride, src, dst, att, ea, We, evals, segmax, E, Cc);
  edge_exp_kernel    <<<cdiv(E, 256), 256, 0, s>>>(evals, dst, segmax, segsum, avals, E);
  edge_scatter_kernel<<<eb, 256, 0, s>>>(l, src, dst, avals, segsum, out, E, Cc);
}

static void lbn_tanh(hipStream_t s, const float* x, float* y, const float* g, const float* b,
                     float* bnsum, float* bnsq, int N, int Cc) {
  lfill(s, bnsum, 0.f, Cc);
  lfill(s, bnsq, 0.f, Cc);
  bn_stats_kernel<<<cdiv(N, 16), 256, 0, s>>>(x, bnsum, bnsq, N, Cc);
  long long tot = (long long)N * Cc;
  bn_apply_tanh_kernel<<<cdiv(tot, 256), 256, 0, s>>>(x, y, bnsum, bnsq, g, b, N, Cc);
}

// =================================================================
extern "C" void kernel_launch(void* const* d_in, const int* in_sizes, int n_in,
                              void* d_out, int out_size, void* d_ws, size_t ws_size,
                              hipStream_t stream) {
  (void)in_sizes; (void)out_size;
  if (n_in < N_LEAVES) return;

  const float* x_op = (const float*)d_in[IN_X_OP];
  const float* x_m  = (const float*)d_in[IN_X_M];
  const float* x_a  = (const float*)d_in[IN_X_A];
  const int* ei_pred_src = (const int*)d_in[IN_EI_PRED];
  const int* ei_pred_dst = ei_pred_src + E_PRED;
  const int* ei_succ_src = (const int*)d_in[IN_EI_SUCC];
  const int* ei_succ_dst = ei_succ_src + E_SUCC;
  const int* ei_om_src = (const int*)d_in[IN_EI_OM_SRC];
  const int* ei_om_dst = (const int*)d_in[IN_EI_OM_DST];
  const float* ea_om = (const float*)d_in[IN_EA_OM];
  const int* ei_ma_src = (const int*)d_in[IN_EI_MA_SRC];
  const int* ei_ma_dst = (const int*)d_in[IN_EI_MA_DST];
  const float* global_attr = (const float*)d_in[IN_GLOBAL];
  const int* batch_op = (const int*)d_in[IN_BATCH_OP];
  const int* batch_m  = (const int*)d_in[IN_BATCH_M];
  const int* batch_a  = (const int*)d_in[IN_BATCH_A];
#define PF(i) ((const float*)d_in[(i)])

  // ---------------- workspace layout (floats) ----------------
  float* w = (float*)d_ws;
  size_t o = 0;
#define WS_ALLOC(name, cnt) float* name = w + o; o += (size_t)(cnt)
  WS_ALLOC(h_op,  (size_t)N_OP * CH);
  WS_ALLOC(h_m,   (size_t)N_M  * CH);
  WS_ALLOC(h_a,   (size_t)N_A  * CH);
  WS_ALLOC(l_buf, (size_t)N_OP * CH);
  WS_ALLOC(r_buf, (size_t)N_OP * CH);
  WS_ALLOC(f_op,  (size_t)N_OP * CH);
  WS_ALLOC(f_m,   (size_t)N_M  * CH);
  WS_ALLOC(f_a,   (size_t)N_A  * CH);
  WS_ALLOC(evals, E_OM);
  WS_ALLOC(avals, E_OM);
  WS_ALLOC(segmax, N_OP);
  WS_ALLOC(segsum, N_OP);
  WS_ALLOC(bnsum, 1024);
  WS_ALLOC(bnsq,  1024);
  WS_ALLOC(rvec,  CH);
  WS_ALLOC(z0,   (size_t)BSZ * (F_GF + 3 * CH));
  WS_ALLOC(tbuf, (size_t)BSZ * 512);
  WS_ALLOC(z1,   (size_t)BSZ * 512);
  WS_ALLOC(z2,   (size_t)BSZ * 512);
  if (ws_size < o * sizeof(float)) return;

  // ---------------- output regions ----------------
  float* out   = (float*)d_out;
  float* o_hop = out;
  float* o_hm  = o_hop + (size_t)N_OP * CH;
  float* o_ha  = o_hm  + (size_t)N_M  * CH;
  float* o_gop = o_ha  + (size_t)N_A  * CH;
  float* o_gm  = o_gop + (size_t)BSZ * CH;
  float* o_ga  = o_gm  + (size_t)BSZ * CH;
  float* o_z   = o_ga  + (size_t)BSZ * CH;

  // ---------------- input projections ----------------
  lgemm(stream, x_op, PF(P_PROJ_OP), PF(P_PROJ_OP + 1), h_op, N_OP, F_OPF, CH, 0, 0);
  lgemm(stream, x_m,  PF(P_PROJ_M),  PF(P_PROJ_M  + 1), h_m,  N_M,  F_MF,  CH, 0, 0);
  lgemm(stream, x_a,  PF(P_PROJ_A),  PF(P_PROJ_A  + 1), h_a,  N_A,  F_AF,  CH, 0, 0);

  // ---------------- GNN layers ----------------
  for (int li = 0; li < NL; ++li) {
    int LB = P_LAYER0 + L_STRIDE * li;
    lfill(stream, f_op, 0.f, (long long)N_OP * CH);
    lfill(stream, f_m,  0.f, (long long)N_M  * CH);
    lfill(stream, f_a,  0.f, (long long)N_A  * CH);

    // pred: op -> op
    lgemm(stream, h_op, PF(LB + L_PRED + 0), PF(LB + L_PRED + 4), l_buf, N_OP, CH, CH, 0, 0);
    lgemm(stream, h_op, PF(LB + L_PRED + 1), PF(LB + L_PRED + 5), r_buf, N_OP, CH, CH, 0, 0);
    lgat(stream, l_buf, r_buf, CH, ei_pred_src, ei_pred_dst, PF(LB + L_PRED + 2),
         nullptr, nullptr, evals, avals, segmax, segsum, f_op, E_PRED, N_OP, CH);

    // succ: op -> op (accumulate into f_op)
    lgemm(stream, h_op, PF(LB + L_SUCC + 0), PF(LB + L_SUCC + 4), l_buf, N_OP, CH, CH, 0, 0);
    lgemm(stream, h_op, PF(LB + L_SUCC + 1), PF(LB + L_SUCC + 5), r_buf, N_OP, CH, CH, 0, 0);
    lgat(stream, l_buf, r_buf, CH, ei_succ_src, ei_succ_dst, PF(LB + L_SUCC + 2),
         nullptr, nullptr, evals, avals, segmax, segsum, f_op, E_SUCC, N_OP, CH);

    // om: op -> m (with edge features)
    lgemm(stream, h_op, PF(LB + L_OM + 1), PF(LB + L_OM + 5), l_buf, N_OP, CH, CH, 0, 0);
    lgemm(stream, h_m,  PF(LB + L_OM + 2), PF(LB + L_OM + 6), r_buf, N_M,  CH, CH, 0, 0);
    lgat(stream, l_buf, r_buf, CH, ei_om_src, ei_om_dst, PF(LB + L_OM + 3),
         ea_om, PF(LB + L_OM + 0), evals, avals, segmax, segsum, f_m, E_OM, N_M, CH);

    // ma: m -> a
    lgemm(stream, h_m, PF(LB + L_MA + 0), PF(LB + L_MA + 4), l_buf, N_M, CH, CH, 0, 0);
    lgemm(stream, h_a, PF(LB + L_MA + 1), PF(LB + L_MA + 5), r_buf, N_A, CH, CH, 0, 0);
    lgat(stream, l_buf, r_buf, CH, ei_ma_src, ei_ma_dst, PF(LB + L_MA + 2),
         nullptr, nullptr, evals, avals, segmax, segsum, f_a, E_MA, N_A, CH);

    // residual + conv biases, then BN + tanh
    bias_res_kernel<<<cdiv((long long)N_OP * CH, 256), 256, 0, stream>>>(
        f_op, h_op, PF(LB + L_PRED + 3), PF(LB + L_SUCC + 3), N_OP, CH);
    lbn_tanh(stream, f_op, h_op, PF(LB + L_BN_OP_G), PF(LB + L_BN_OP_B), bnsum, bnsq, N_OP, CH);

    bias_res_kernel<<<cdiv((long long)N_M * CH, 256), 256, 0, stream>>>(
        f_m, h_m, PF(LB + L_OM + 4), nullptr, N_M, CH);
    lbn_tanh(stream, f_m, h_m, PF(LB + L_BN_M_G), PF(LB + L_BN_M_B), bnsum, bnsq, N_M, CH);

    bias_res_kernel<<<cdiv((long long)N_A * CH, 256), 256, 0, stream>>>(
        f_a, h_a, PF(LB + L_MA + 3), nullptr, N_A, CH);
    lbn_tanh(stream, f_a, h_a, PF(LB + L_BN_A_G), PF(LB + L_BN_A_B), bnsum, bnsq, N_A, CH);
  }

  // ---------------- StateMixer: node -> global GATv2 pooling ----------------
  struct PoolSpec { const float* h; const int* batch; int n; int conv; int tok; float* g; };
  PoolSpec pools[3] = {
    { h_op, batch_op, N_OP, P_CONV_OP, P_TOK_OP, o_gop },
    { h_m,  batch_m,  N_M,  P_CONV_M,  P_TOK_M,  o_gm  },
    { h_a,  batch_a,  N_A,  P_CONV_A,  P_TOK_A,  o_ga  },
  };
  for (int pi = 0; pi < 3; ++pi) {
    PoolSpec& P = pools[pi];
    lfill(stream, P.g, 0.f, (long long)BSZ * CH);
    lgemm(stream, P.h, PF(P.conv + 0), PF(P.conv + 4), l_buf, P.n, CH, CH, 0, 0);
    lgemm(stream, PF(P.tok), PF(P.conv + 1), PF(P.conv + 5), rvec, 1, CH, CH, 0, 0);
    lgat(stream, l_buf, rvec, /*rstride=*/0, /*src=*/nullptr, P.batch, PF(P.conv + 2),
         nullptr, nullptr, evals, avals, segmax, segsum, P.g, P.n, BSZ, CH);
    bias_res_kernel<<<cdiv((long long)BSZ * CH, 256), 256, 0, stream>>>(
        P.g, nullptr, PF(P.conv + 3), nullptr, BSZ, CH);
  }

  // ---------------- mixer MLP ----------------
  const int ZIN = F_GF + 3 * CH;  // 784
  concat_z_kernel<<<cdiv((long long)BSZ * ZIN, 256), 256, 0, stream>>>(
      global_attr, o_gop, o_gm, o_ga, z0, BSZ, F_GF, CH);

  // mix0: z1 = z0@Wp+bp + tanh(z0@W1+b1)@W2+b2 ; BN ; tanh
  lgemm(stream, z0, PF(P_MIX0 + 0), PF(P_MIX0 + 3), tbuf, BSZ, ZIN, 512, 1, 0);
  lgemm(stream, tbuf, PF(P_MIX0 + 1), PF(P_MIX0 + 4), z1, BSZ, 512, 512, 0, 0);
  lgemm(stream, z0, PF(P_MIX0 + 2), PF(P_MIX0 + 5), z1, BSZ, ZIN, 512, 0, 1);
  lbn_tanh(stream, z1, z1, PF(P_MIXBN0 + 1), PF(P_MIXBN0 + 0), bnsum, bnsq, BSZ, 512);

  // mix1: z2 = z1 + tanh(z1@W1+b1)@W2+b2 ; BN ; tanh
  lgemm(stream, z1, PF(P_MIX1 + 0), PF(P_MIX1 + 2), tbuf, BSZ, 512, 512, 1, 0);
  lgemm(stream, tbuf, PF(P_MIX1 + 1), PF(P_MIX1 + 3), z2, BSZ, 512, 512, 0, 0);
  add_inplace_kernel<<<cdiv((long long)BSZ * 512, 256), 256, 0, stream>>>(
      z2, z1, (long long)BSZ * 512);
  lbn_tanh(stream, z2, z2, PF(P_MIXBN1 + 1), PF(P_MIXBN1 + 0), bnsum, bnsq, BSZ, 512);

  // mix2: z = z2@Wp+bp + tanh(z2@W1+b1)@W2+b2  (written straight into d_out)
  lgemm(stream, z2, PF(P_MIX2 + 0), PF(P_MIX2 + 3), tbuf, BSZ, 512, 256, 1, 0);
  lgemm(stream, tbuf, PF(P_MIX2 + 1), PF(P_MIX2 + 4), o_z, BSZ, 256, 256, 0, 0);
  lgemm(stream, z2, PF(P_MIX2 + 2), PF(P_MIX2 + 5), o_z, BSZ, 512, 256, 0, 1);

  // ---------------- final node-feature outputs ----------------
  copy_kernel<<<cdiv((long long)N_OP * CH, 256), 256, 0, stream>>>(o_hop, h_op, (long long)N_OP * CH);
  copy_kernel<<<cdiv((long long)N_M  * CH, 256), 256, 0, stream>>>(o_hm,  h_m,  (long long)N_M  * CH);
  copy_kernel<<<cdiv((long long)N_A  * CH, 256), 256, 0, stream>>>(o_ha,  h_a,  (long long)N_A  * CH);
}
